// GNN_47682726921133
// MI455X (gfx1250) — matrix operand order repaired and checked by reading, not compiled, not run on hardware
//
#include <hip/hip_runtime.h>
#include <hip/hip_bf16.h>
#include <stdint.h>

// ---------------------------------------------------------------------------
// GNN layer for MI455X (gfx1250), wave32 + WMMA bf16.
//
// Phases (all on `stream`, stream-ordered so no extra sync needed):
//   0) n_f -> bf16 table; pack 5 weight matrices into WMMA-B fragment order
//   1) fused edge MLP + attention logit (WMMA bf16, e_f never materialized)
//   2) per-dst softmax (encoded-uint atomicMax, expf, atomicAdd denom)
//   3) z scatter: z[dst] += alpha * n_f[src]  (f32 atomics, L2-resident)
//   4) z -> bf16; node MLP (WMMA bf16) with relu -> d_out
//
// The GEMM hot loop uses sched_group_barrier (0x20 = VMEM read, 0x8 = WMMA)
// to force "all 18 loads, then 8 WMMAs" per K-tile so the waitcnt pass emits
// partial s_wait_loadcnt and WMMAs overlap in-flight B loads (the default
// scheduler sank each load pair to its WMMA -> s_wait_loadcnt 0 every time).
//
// Workspace: ~186 MB (n_bf 41MB, 5x packed W 21MB, z 82MB, z_bf 41MB, misc).
// ---------------------------------------------------------------------------

typedef __attribute__((ext_vector_type(16))) __bf16 v16bf;
typedef __attribute__((ext_vector_type(8)))  float  v8f;
typedef __attribute__((ext_vector_type(4)))  unsigned int v4u;

#define D_FEAT 1024
#define KT_TILES 64   // K = 2048 -> 64 tiles of 32
#define NT_TILES 64   // N = 1024 -> 64 tiles of 16

union FragAB { v16bf v; v4u u[2]; };

__device__ __forceinline__ unsigned fenc(float x) {
  unsigned u = __float_as_uint(x);
  return (u & 0x80000000u) ? ~u : (u | 0x80000000u);   // order-preserving f32->u32
}
__device__ __forceinline__ float fdec(unsigned e) {
  unsigned u = (e & 0x80000000u) ? (e ^ 0x80000000u) : ~e;
  return __uint_as_float(u);
}
__device__ __forceinline__ unsigned short f2bf(float x) {
  __bf16 b = (__bf16)x;
  return *reinterpret_cast<unsigned short*>(&b);
}

// ---------------- utility kernels ----------------

__global__ void k_f32_to_bf16(const float* __restrict__ in,
                              unsigned short* __restrict__ out, size_t n) {
  size_t i = (size_t)blockIdx.x * blockDim.x + threadIdx.x;
  if (i < n) out[i] = f2bf(in[i]);
}

__global__ void k_fill_u32(unsigned* __restrict__ p, unsigned v, size_t n) {
  size_t i = (size_t)blockIdx.x * blockDim.x + threadIdx.x;
  if (i < n) p[i] = v;
}

// Pack W[2048,1024] (row=K, col=N) into WMMA-B fragment order:
// Wp[nt][kt][lane][i]  with  K = kt*32 + (i>>3)*16 + (lane>>4)*8 + (i&7),  N = nt*16 + (lane&15)
// so each lane's 16 bf16 (32B) of a B fragment are contiguous.
__global__ void k_pack_w(const float* __restrict__ W,
                         unsigned short* __restrict__ Wp) {
  size_t t = (size_t)blockIdx.x * blockDim.x + threadIdx.x;  // < 2^21
  int i    = (int)(t & 15);
  int lane = (int)((t >> 4) & 31);
  int kt   = (int)((t >> 9) & 63);
  int nt   = (int)(t >> 15);
  int K = kt * 32 + (i >> 3) * 16 + (lane >> 4) * 8 + (i & 7);
  int N = nt * 16 + (lane & 15);
  Wp[t] = f2bf(W[(size_t)K * D_FEAT + N]);
}

// ---------------- phase 1: fused edge MLP + attention logit ----------------
// One block = 16 edges. 8 waves; wave w owns n-tiles [w*8, w*8+8).
// A fragment (16x32 bf16): lane holds row M=lane&15 of the gathered
// concat(n_f[src], n_f[dst]); K chunks per ISA layout:
//   lanes 0-15: K = kt*32 + {0..7} u {16..23};  lanes 16-31: +8.
__global__ void __launch_bounds__(256)
k_edge_logits(const unsigned short* __restrict__ n_bf,
              const int* __restrict__ src, const int* __restrict__ dst,
              const unsigned short* __restrict__ Wp,
              const float* __restrict__ bias, const float* __restrict__ watt,
              const float* __restrict__ b_att,
              float* __restrict__ a_out) {
  __shared__ float s_att[16];
  const int tid = threadIdx.x;
  const int wave = tid >> 5, lane = tid & 31;
  const int group = blockIdx.x;
  if (tid < 16) s_att[tid] = 0.f;
  __syncthreads();

  const int mrow = lane & 15;          // edge row within group (A) / column (C)
  const int hi   = lane >> 4;          // half-wave selector
  const int e    = group * 16 + mrow;
  const size_t srow = (size_t)src[e] * D_FEAT;
  const size_t drow = (size_t)dst[e] * D_FEAT;
  __builtin_prefetch(n_bf + drow, 0, 1);   // warm dst row (global_prefetch)

  v8f c[8] = {};

#pragma unroll 1
  for (int kt = 0; kt < KT_TILES; ++kt) {
    const size_t rbase = (kt < 32) ? srow : drow;           // concat(src,dst)
    const int f = ((kt & 31) << 5) + hi * 8;                // feature offset
    const v4u* pa = reinterpret_cast<const v4u*>(n_bf + rbase + f);
    const unsigned short* wb =
        Wp + (((size_t)(wave * 8) * KT_TILES + kt) * 32 + lane) * 16;

    // batch all loads first; scheduling directives below keep them ahead of
    // the WMMAs so waits are partial (overlap matrix ops with loads)
    FragAB a;
    a.u[0] = pa[0];        // K = f .. f+7
    a.u[1] = pa[2];        // K = f+16 .. f+23
    FragAB b[8];
#pragma unroll
    for (int j = 0; j < 8; ++j) {
      const v4u* pb =
          reinterpret_cast<const v4u*>(wb + (size_t)j * KT_TILES * 32 * 16);
      b[j].u[0] = pb[0];
      b[j].u[1] = pb[1];
    }
#pragma unroll
    for (int j = 0; j < 8; ++j) {
      c[j] = __builtin_amdgcn_wmma_f32_16x16x32_bf16(
          false, a.v, false, b[j].v, (short)0, c[j], false, false);
    }
    // schedule pin: 18 VMEM reads (A pair + 8 B pairs) then 8 WMMAs
    __builtin_amdgcn_sched_group_barrier(0x0020, 18, 0);  // VMEM read
    __builtin_amdgcn_sched_group_barrier(0x0008, 8, 0);   // MFMA/WMMA
  }

  // relu(h + bias) dot W_att, reduced over this wave's 128 hidden columns.
  // C layout: lane covers column n = lane&15 of tile; rows M = hi*8 + r.
  float attp[8] = {0.f, 0.f, 0.f, 0.f, 0.f, 0.f, 0.f, 0.f};
  const int col = lane & 15;
#pragma unroll
  for (int j = 0; j < 8; ++j) {
    const int n = (wave * 8 + j) * 16 + col;
    const float bn = bias[n];
    const float wn = watt[n];
#pragma unroll
    for (int r = 0; r < 8; ++r) {
      float h = c[j][r] + bn;
      h = h > 0.f ? h : 0.f;
      attp[r] += h * wn;
    }
  }
  // sum over the 16 columns (lanes within each half keep distinct row sets)
#pragma unroll
  for (int m = 1; m < 16; m <<= 1)
#pragma unroll
    for (int r = 0; r < 8; ++r) attp[r] += __shfl_xor(attp[r], m, 32);
  if (col == 0) {                      // lane 0 (rows 0-7) and lane 16 (rows 8-15)
#pragma unroll
    for (int r = 0; r < 8; ++r) atomicAdd(&s_att[hi * 8 + r], attp[r]);
  }
  __syncthreads();
  if (tid < 16) a_out[group * 16 + tid] = s_att[tid] + b_att[0];
}

// ---------------- phase 2: softmax pieces ----------------

__global__ void k_seg_max(const float* __restrict__ a, const int* __restrict__ dst,
                          unsigned* __restrict__ maxenc, int E) {
  int e = blockIdx.x * blockDim.x + threadIdx.x;
  if (e < E) atomicMax(&maxenc[dst[e]], fenc(a[e]));
}

__global__ void k_exp_denom(const float* __restrict__ a, const int* __restrict__ dst,
                            const unsigned* __restrict__ maxenc,
                            float* __restrict__ ex, float* __restrict__ denom, int E) {
  int e = blockIdx.x * blockDim.x + threadIdx.x;
  if (e < E) {
    const int d = dst[e];
    const float v = __expf(a[e] - fdec(maxenc[d]));
    ex[e] = v;
    atomicAdd(&denom[d], v);
  }
}

// one block (256 threads) per edge: z[dst] += alpha * n_f[src]
__global__ void k_scatter_z(const float* __restrict__ n_f,
                            const int* __restrict__ src, const int* __restrict__ dst,
                            const float* __restrict__ ex, const float* __restrict__ denom,
                            float* __restrict__ z) {
  const int e = blockIdx.x;
  const int d = dst[e], s = src[e];
  const float alpha = ex[e] / denom[d];
  const int i0 = threadIdx.x * 4;
  const float4 v = *reinterpret_cast<const float4*>(n_f + (size_t)s * D_FEAT + i0);
  float* zp = z + (size_t)d * D_FEAT + i0;
  atomicAdd(zp + 0, alpha * v.x);
  atomicAdd(zp + 1, alpha * v.y);
  atomicAdd(zp + 2, alpha * v.z);
  atomicAdd(zp + 3, alpha * v.w);
}

// ---------------- phase 4: node MLP ----------------

__global__ void __launch_bounds__(256)
k_node_mlp(const unsigned short* __restrict__ n_bf,
           const unsigned short* __restrict__ z_bf,
           const unsigned short* __restrict__ Wp,
           const float* __restrict__ bias,
           float* __restrict__ out, int rowbase, int rowlimit) {
  const int tid = threadIdx.x;
  const int wave = tid >> 5, lane = tid & 31;
  const int group = blockIdx.x;
  const int mrow = lane & 15, hi = lane >> 4, col = lane & 15;

  int rowA = rowbase + group * 16 + mrow;
  if (rowA > rowlimit - 1) rowA = rowlimit - 1;        // clamp; stores masked
  const size_t arow = (size_t)rowA * D_FEAT;

  v8f c[8] = {};

#pragma unroll 1
  for (int kt = 0; kt < KT_TILES; ++kt) {
    const unsigned short* tab = (kt < 32) ? n_bf : z_bf;  // concat(n_f, z)
    const int f = ((kt & 31) << 5) + hi * 8;
    const v4u* pa = reinterpret_cast<const v4u*>(tab + arow + f);
    const unsigned short* wb =
        Wp + (((size_t)(wave * 8) * KT_TILES + kt) * 32 + lane) * 16;

    FragAB a;
    a.u[0] = pa[0];
    a.u[1] = pa[2];
    FragAB b[8];
#pragma unroll
    for (int j = 0; j < 8; ++j) {
      const v4u* pb =
          reinterpret_cast<const v4u*>(wb + (size_t)j * KT_TILES * 32 * 16);
      b[j].u[0] = pb[0];
      b[j].u[1] = pb[1];
    }
#pragma unroll
    for (int j = 0; j < 8; ++j) {
      c[j] = __builtin_amdgcn_wmma_f32_16x16x32_bf16(
          false, a.v, false, b[j].v, (short)0, c[j], false, false);
    }
    __builtin_amdgcn_sched_group_barrier(0x0020, 18, 0);  // VMEM read
    __builtin_amdgcn_sched_group_barrier(0x0008, 8, 0);   // MFMA/WMMA
  }

#pragma unroll
  for (int j = 0; j < 8; ++j) {
    const int n = (wave * 8 + j) * 16 + col;
    const float bn = bias[n];
#pragma unroll
    for (int r = 0; r < 8; ++r) {
      const int m = hi * 8 + r;
      const int row = rowbase + group * 16 + m;
      if (row < rowlimit) {
        float h = c[j][r] + bn;
        out[(size_t)row * D_FEAT + n] = h > 0.f ? h : 0.f;
      }
    }
  }
}

// ---------------- host launcher ----------------

static inline size_t pad256(size_t x) { return (x + 255) & ~(size_t)255; }

extern "C" void kernel_launch(void* const* d_in, const int* in_sizes, int n_in,
                              void* d_out, int out_size, void* d_ws, size_t ws_size,
                              hipStream_t stream) {
  (void)n_in; (void)out_size; (void)ws_size;
  const float* n_f   = (const float*)d_in[0];
  const int* src_hh  = (const int*)d_in[1];
  const int* dst_hh  = (const int*)d_in[2];
  const int* src_oo  = (const int*)d_in[3];
  const int* dst_oo  = (const int*)d_in[4];
  const int* src_ho  = (const int*)d_in[5];
  const int* dst_ho  = (const int*)d_in[6];
  const float* W_hh  = (const float*)d_in[7];
  const float* b_hh  = (const float*)d_in[8];
  const float* W_oo  = (const float*)d_in[9];
  const float* b_oo  = (const float*)d_in[10];
  const float* W_ho  = (const float*)d_in[11];
  const float* b_ho  = (const float*)d_in[12];
  const float* W_att = (const float*)d_in[13];
  const float* b_att = (const float*)d_in[14];
  const float* W_hn  = (const float*)d_in[15];
  const float* b_hn  = (const float*)d_in[16];
  const float* W_on  = (const float*)d_in[17];
  const float* b_on  = (const float*)d_in[18];

  const int E_hh = in_sizes[1], E_oo = in_sizes[3], E_ho = in_sizes[5];
  const size_t ND = (size_t)in_sizes[0];
  const int N = (int)(ND / D_FEAT);
  const int N_H = 5000;  // fixed by the reference problem
  const int E = E_hh + E_oo + E_ho;
  const size_t WPK = (size_t)NT_TILES * KT_TILES * 32 * 16;  // 2^21 bf16 / weight

  // workspace carve-up
  char* ws = (char*)d_ws;
  unsigned short* n_bf  = (unsigned short*)ws;  ws += pad256(ND * 2);
  unsigned short* Wp_hh = (unsigned short*)ws;  ws += pad256(WPK * 2);
  unsigned short* Wp_oo = (unsigned short*)ws;  ws += pad256(WPK * 2);
  unsigned short* Wp_ho = (unsigned short*)ws;  ws += pad256(WPK * 2);
  unsigned short* Wp_hn = (unsigned short*)ws;  ws += pad256(WPK * 2);
  unsigned short* Wp_on = (unsigned short*)ws;  ws += pad256(WPK * 2);
  float*    a_buf  = (float*)ws;                ws += pad256((size_t)E * 4);
  float*    ex_buf = (float*)ws;                ws += pad256((size_t)E * 4);
  unsigned* maxenc = (unsigned*)ws;             ws += pad256((size_t)N * 4);
  float*    denom  = (float*)ws;                ws += pad256((size_t)N * 4);
  float*    z      = (float*)ws;                ws += pad256(ND * 4);
  unsigned short* z_bf = (unsigned short*)ws;   ws += pad256(ND * 2);

  // phase 0: precision convert + weight fragment packing
  k_f32_to_bf16<<<(unsigned)((ND + 255) / 256), 256, 0, stream>>>(n_f, n_bf, ND);
  const unsigned packGrid = (unsigned)(WPK / 256);
  k_pack_w<<<packGrid, 256, 0, stream>>>(W_hh, Wp_hh);
  k_pack_w<<<packGrid, 256, 0, stream>>>(W_oo, Wp_oo);
  k_pack_w<<<packGrid, 256, 0, stream>>>(W_ho, Wp_ho);
  k_pack_w<<<packGrid, 256, 0, stream>>>(W_hn, Wp_hn);
  k_pack_w<<<packGrid, 256, 0, stream>>>(W_on, Wp_on);

  // init softmax/aggregation state
  k_fill_u32<<<(unsigned)((ND + 255) / 256), 256, 0, stream>>>((unsigned*)z, 0u, ND);
  k_fill_u32<<<(unsigned)((N + 255) / 256), 256, 0, stream>>>((unsigned*)denom, 0u, (size_t)N);
  k_fill_u32<<<(unsigned)((N + 255) / 256), 256, 0, stream>>>(maxenc, 0u, (size_t)N);

  // phase 1: fused edge MLP + attention logits (a_buf = [hh | oo | ho])
  k_edge_logits<<<(unsigned)(E_hh / 16), 256, 0, stream>>>(
      n_bf, src_hh, dst_hh, Wp_hh, b_hh, W_att, b_att, a_buf);
  k_edge_logits<<<(unsigned)(E_oo / 16), 256, 0, stream>>>(
      n_bf, src_oo, dst_oo, Wp_oo, b_oo, W_att, b_att, a_buf + E_hh);
  k_edge_logits<<<(unsigned)(E_ho / 16), 256, 0, stream>>>(
      n_bf, src_ho, dst_ho, Wp_ho, b_ho, W_att, b_att, a_buf + E_hh + E_oo);

  // phase 2: per-dst softmax
  k_seg_max<<<(unsigned)((E_hh + 255) / 256), 256, 0, stream>>>(a_buf, dst_hh, maxenc, E_hh);
  k_seg_max<<<(unsigned)((E_oo + 255) / 256), 256, 0, stream>>>(a_buf + E_hh, dst_oo, maxenc, E_oo);
  k_seg_max<<<(unsigned)((E_ho + 255) / 256), 256, 0, stream>>>(a_buf + E_hh + E_oo, dst_ho, maxenc, E_ho);
  k_exp_denom<<<(unsigned)((E_hh + 255) / 256), 256, 0, stream>>>(a_buf, dst_hh, maxenc, ex_buf, denom, E_hh);
  k_exp_denom<<<(unsigned)((E_oo + 255) / 256), 256, 0, stream>>>(a_buf + E_hh, dst_oo, maxenc, ex_buf + E_hh, denom, E_oo);
  k_exp_denom<<<(unsigned)((E_ho + 255) / 256), 256, 0, stream>>>(a_buf + E_hh + E_oo, dst_ho, maxenc, ex_buf + E_hh + E_oo, denom, E_ho);

  // phase 3: z[dst] += alpha * n_f[src]
  k_scatter_z<<<(unsigned)E_hh, 256, 0, stream>>>(n_f, src_hh, dst_hh, ex_buf, denom, z);
  k_scatter_z<<<(unsigned)E_oo, 256, 0, stream>>>(n_f, src_oo, dst_oo, ex_buf + E_hh, denom, z);
  k_scatter_z<<<(unsigned)E_ho, 256, 0, stream>>>(n_f, src_ho, dst_ho, ex_buf + E_hh + E_oo, denom, z);

  // phase 4: node MLPs
  k_f32_to_bf16<<<(unsigned)((ND + 255) / 256), 256, 0, stream>>>(z, z_bf, ND);
  const int groups_h = (N_H + 15) / 16;
  const int groups_o = (N - N_H + 15) / 16;
  k_node_mlp<<<(unsigned)groups_h, 256, 0, stream>>>(n_bf, z_bf, Wp_hn, b_hn,
                                                     (float*)d_out, 0, N_H);
  k_node_mlp<<<(unsigned)groups_o, 256, 0, stream>>>(n_bf, z_bf, Wp_on, b_on,
                                                     (float*)d_out, N_H, N);
}
